// MultiHeadAttention_42752104464925
// MI455X (gfx1250) — matrix operand order repaired and checked by reading, not compiled, not run on hardware
//
#include <hip/hip_runtime.h>
#include <hip/hip_bf16.h>
#include <math.h>
#include <stdint.h>

// ---------------------------------------------------------------------------
// MI455X (gfx1250) relative-position MHA, bf16 WMMA everywhere.
//   B=4, C=768, T=1024, H=12, KC=64, WIN=4
// ---------------------------------------------------------------------------

typedef __attribute__((ext_vector_type(16))) __bf16 v16bf;
typedef __attribute__((ext_vector_type(8)))  float  v8f;

#define B_  4
#define C_  768
#define T_  1024
#define H_  12
#define KC_ 64

union Frag {           // WMMA 16-bit operand: 8 VGPRs = 8 packed bf16 pairs
  v16bf bf;
  unsigned int u[8];
};

union Pk {
  unsigned int u;
  __bf16 h[2];
};

static __device__ __forceinline__ v8f wmma_bf16(v16bf a, v16bf b, v8f c) {
  return __builtin_amdgcn_wmma_f32_16x16x32_bf16(false, a, false, b, (short)0, c,
                                                 false, false);
}

static __device__ __forceinline__ v8f zero8() {
  v8f z;
#pragma unroll
  for (int i = 0; i < 8; ++i) z[i] = 0.f;
  return z;
}

static __device__ __forceinline__ unsigned int pk_bf16(float a, float b) {
  Pk r;
  r.h[0] = (__bf16)a;   // lowers to v_cvt_pk_bf16_f32
  r.h[1] = (__bf16)b;
  return r.u;
}

// ===========================================================================
// Kernel 1: fused Q/K/V 1x1-conv projections.
//   out[o,t] = W[o,:]·src[:,t] + bias[o]   (q additionally scaled by 1/8)
//   stored bf16 in [B,H,T,KC] layout for the attention kernel.
// Block: 128 threads = 4 waves in a 2x2 grid, block tile 64(o) x 64(t).
// ===========================================================================
__global__ __launch_bounds__(128) void qkv_proj_kernel(
    const float* __restrict__ x, const float* __restrict__ cin,
    const float* __restrict__ Wq, const float* __restrict__ bq,
    const float* __restrict__ Wk, const float* __restrict__ bk,
    const float* __restrict__ Wv, const float* __restrict__ bv,
    __bf16* __restrict__ qd, __bf16* __restrict__ kd, __bf16* __restrict__ vd) {
  __shared__ __bf16 Wl[64][36];    // [o][c]  A-major: pairs along c contiguous
  __shared__ __bf16 XlT[64][40];   // [t][c]  transposed so K-pairs contiguous

  const int tTile = blockIdx.x;              // T/64 = 16
  const int oTile = blockIdx.y;              // C/64 = 12
  const int bz    = blockIdx.z;              // B*3  = 12
  const int b = bz / 3, m = bz % 3;

  const float* W    = (m == 0) ? Wq : (m == 1) ? Wk : Wv;
  const float* bias = (m == 0) ? bq : (m == 1) ? bk : bv;
  const float* src  = ((m == 0) ? x : cin) + (size_t)b * C_ * T_;
  __bf16*      dst  = (m == 0) ? qd : (m == 1) ? kd : vd;
  const float scale = (m == 0) ? 0.125f : 1.0f;  // 1/sqrt(KC) folded into q

  const int tid = threadIdx.x, lane = tid & 31, wid = tid >> 5;
  const int hl = lane >> 4, ln = lane & 15;
  const int wm = wid >> 1, wn = wid & 1;
  const int oBase = oTile * 64, tBase = tTile * 64;

  v8f acc[2][2];
#pragma unroll
  for (int i = 0; i < 2; ++i)
#pragma unroll
    for (int j = 0; j < 2; ++j) acc[i][j] = zero8();

#pragma unroll 1
  for (int kb = 0; kb < C_ / 32; ++kb) {
    // W tile: pack fp32 pairs -> one b32 store each
    for (int i = tid; i < 64 * 16; i += 128) {
      int r = i >> 4, p = i & 15;
      const float* wp = &W[(size_t)(oBase + r) * C_ + kb * 32 + 2 * p];
      ((unsigned int*)&Wl[r][0])[p] = pk_bf16(wp[0], wp[1]);
    }
    // X tile transposed: consecutive tid -> consecutive t (coalesced reads)
    for (int i = tid; i < 32 * 64; i += 128) {
      int r = i >> 6, t = i & 63;
      XlT[t][r] = (__bf16)src[(size_t)(kb * 32 + r) * T_ + tBase + t];
    }
    __syncthreads();
    if (kb + 1 < C_ / 32) {  // global_prefetch_b8 of next tiles
      __builtin_prefetch(&W[(size_t)(oBase + (tid >> 1)) * C_ + (kb + 1) * 32 + (tid & 1) * 16], 0, 0);
      __builtin_prefetch(&src[(size_t)((kb + 1) * 32 + (tid >> 2)) * T_ + tBase + (tid & 3) * 16], 0, 0);
    }

    Frag afr[2], bfr[2];
#pragma unroll
    for (int ms = 0; ms < 2; ++ms) {      // A 16x32: rows on lanes 0..15
      int mr = wm * 32 + ms * 16 + ln;
      int kg = hl * 16;
#pragma unroll
      for (int vv = 0; vv < 8; ++vv) {
        int k0 = kg + 2 * vv;
        afr[ms].u[vv] = *(const unsigned int*)&Wl[mr][k0];
      }
    }
#pragma unroll
    for (int ns = 0; ns < 2; ++ns) {      // B 32x16: cols on lanes 0..15
      int nc = wn * 32 + ns * 16 + ln;
      int k2 = hl * 16;
#pragma unroll
      for (int vv = 0; vv < 8; ++vv)
        bfr[ns].u[vv] = *(const unsigned int*)&XlT[nc][k2 + 2 * vv];
    }
#pragma unroll
    for (int ms = 0; ms < 2; ++ms)
#pragma unroll
      for (int ns = 0; ns < 2; ++ns)
        acc[ms][ns] = wmma_bf16(afr[ms].bf, bfr[ns].bf, acc[ms][ns]);
    __syncthreads();
  }

  // Epilogue: 8 consecutive o per (lane,half) -> 4 packed b32 stores
#pragma unroll
  for (int ms = 0; ms < 2; ++ms)
#pragma unroll
    for (int ns = 0; ns < 2; ++ns) {
      int o0 = oBase + wm * 32 + ms * 16 + hl * 8;
      int t  = tBase + wn * 32 + ns * 16 + ln;
      int h = o0 >> 6, d0 = o0 & 63;
      unsigned int* dp =
          (unsigned int*)&dst[((size_t)(b * H_ + h) * T_ + t) * KC_ + d0];
#pragma unroll
      for (int p = 0; p < 4; ++p) {
        float va = (acc[ms][ns][2 * p]     + bias[o0 + 2 * p])     * scale;
        float vb = (acc[ms][ns][2 * p + 1] + bias[o0 + 2 * p + 1]) * scale;
        dp[p] = pk_bf16(va, vb);
      }
    }
}

// ===========================================================================
// Kernel 2: flash attention with banded relative-position terms.
// Grid: (T/64, B*H). Block 128 threads = 4 waves; each wave owns 16 q rows.
// K tile staged with global_load_async_to_lds_b128 (ASYNCcnt path);
// V tile staged transposed so P@V B-fragments are packed b32 loads.
// Key-block loop kept ROLLED (one hot body in I$).
// ===========================================================================
__global__ __launch_bounds__(128) void attn_kernel(
    const __bf16* __restrict__ q, const __bf16* __restrict__ k,
    const __bf16* __restrict__ v, const float* __restrict__ relk,
    const float* __restrict__ relv_g, __bf16* __restrict__ att) {
  __shared__ __bf16 Kt[64][72];        // [s][d]   rows 144B (16B-aligned)
  __shared__ __bf16 Vt[64][72];        // [d][s]   transposed
  __shared__ __bf16 Ps[4][16][72];     // per-wave P tile [row][s_local]
  __shared__ float  relb[64][12];      // q_scaled[t]·relk[j], j=0..8
  __shared__ float  relv[9][64];

  const int qTile = blockIdx.x;
  const int bh    = blockIdx.y;
  const int qBase = qTile * 64;
  const int tid = threadIdx.x, lane = tid & 31, wid = tid >> 5;
  const int hl = lane >> 4, ln = lane & 15;

  const __bf16* qp = q + (size_t)bh * T_ * KC_;
  const __bf16* kp = k + (size_t)bh * T_ * KC_;
  const __bf16* vp = v + (size_t)bh * T_ * KC_;

  for (int i = tid; i < 9 * 64; i += 128) relv[i >> 6][i & 63] = relv_g[i];
#pragma unroll 1
  for (int i = tid; i < 64 * 9; i += 128) {
    int tl = i / 9, j = i % 9;
    const __bf16* qr = qp + (size_t)(qBase + tl) * KC_;
    float s = 0.f;
    for (int d = 0; d < KC_; ++d) s += (float)qr[d] * relk[j * KC_ + d];
    relb[tl][j] = s;
  }

  // Q fragments (held for the whole pass): packed b32 global loads
  Frag aq0, aq1;
  {
    const __bf16* qr = qp + (size_t)(qBase + wid * 16 + ln) * KC_;
    int kg = hl * 16;
#pragma unroll
    for (int vv = 0; vv < 8; ++vv) {
      int k0 = kg + 2 * vv;
      aq0.u[vv] = *(const unsigned int*)(qr + k0);
      aq1.u[vv] = *(const unsigned int*)(qr + 32 + k0);
    }
  }

  v8f accO[4];
#pragma unroll
  for (int i = 0; i < 4; ++i) accO[i] = zero8();
  float mrun[8], lrun[8];
#pragma unroll
  for (int i = 0; i < 8; ++i) { mrun[i] = -1e30f; lrun[i] = 0.f; }

  __syncthreads();  // relb / relv ready

#pragma unroll 1
  for (int kb = 0; kb < T_ / 64; ++kb) {
    const int sBase = kb * 64;
    __syncthreads();  // previous iteration done reading Kt/Vt
    // ---- K tile: async DMA global->LDS, 16B per lane per issue ----
#pragma unroll
    for (int c0 = tid; c0 < 512; c0 += 128) {
      int r = c0 >> 3, ch = (c0 & 7) * 8;
      unsigned long long ga =
          (unsigned long long)(uintptr_t)(kp + (size_t)(sBase + r) * KC_ + ch);
      unsigned int la = (unsigned int)(uintptr_t)&Kt[r][ch];
      asm volatile("global_load_async_to_lds_b128 %0, %1, off"
                   :: "v"(la), "v"(ga) : "memory");
    }
    // ---- V tile: manual transpose [s][d] -> [d][s] ----
    {
      const unsigned int* vsrc = (const unsigned int*)(vp + (size_t)sBase * KC_);
#pragma unroll 1
      for (int i = tid; i < 64 * 32; i += 128) {
        int s = i >> 5, dp = (i & 31) * 2;
        Pk w; w.u = vsrc[i];
        Vt[dp][s]     = w.h[0];
        Vt[dp + 1][s] = w.h[1];
      }
    }
    asm volatile("s_wait_asynccnt 0x0" ::: "memory");
    __syncthreads();
    if (kb + 1 < T_ / 64) {
      __builtin_prefetch(vp + (size_t)(sBase + 64) * KC_ + tid * 32, 0, 0);
    }

    // ---- scores: 16 rows x 64 cols per wave (8 WMMA) ----
    v8f accS[4];
#pragma unroll
    for (int n = 0; n < 4; ++n) {
      Frag b0, b1;
      int nn = n * 16 + ln;   // column s_local
      int k2 = hl * 16;
#pragma unroll
      for (int vv = 0; vv < 8; ++vv) {
        int kk = k2 + 2 * vv;          // K-dim = d, pairs contiguous
        b0.u[vv] = *(const unsigned int*)&Kt[nn][kk];
        b1.u[vv] = *(const unsigned int*)&Kt[nn][32 + kk];
      }
      v8f z = zero8();
      z = wmma_bf16(aq0.bf, b0.bf, z);
      z = wmma_bf16(aq1.bf, b1.bf, z);
      accS[n] = z;
    }

    // ---- banded rel-k bias ----
#pragma unroll
    for (int n = 0; n < 4; ++n)
#pragma unroll
      for (int vv = 0; vv < 8; ++vv) {
        int ml = wid * 16 + hl * 8 + vv;     // local q row 0..63
        int dd = (sBase + n * 16 + ln) - (qBase + ml) + 4;
        if (dd >= 0 && dd <= 8) accS[n][vv] += relb[ml][dd];
      }

    // ---- online softmax (rows striped: half hl, vgpr vv) ----
#pragma unroll
    for (int vv = 0; vv < 8; ++vv) {
      float mx = fmaxf(fmaxf(accS[0][vv], accS[1][vv]),
                       fmaxf(accS[2][vv], accS[3][vv]));
#pragma unroll
      for (int off = 1; off < 16; off <<= 1) mx = fmaxf(mx, __shfl_xor(mx, off, 32));
      float mnew  = fmaxf(mrun[vv], mx);
      float alpha = __expf(mrun[vv] - mnew);
      int rw = hl * 8 + vv;
      float ps = 0.f;
#pragma unroll
      for (int n = 0; n < 4; ++n) {
        float p = __expf(accS[n][vv] - mnew);
        Ps[wid][rw][n * 16 + ln] = (__bf16)p;
        ps += p;
      }
#pragma unroll
      for (int off = 1; off < 16; off <<= 1) ps += __shfl_xor(ps, off, 32);
      lrun[vv] = lrun[vv] * alpha + ps;
      mrun[vv] = mnew;
#pragma unroll
      for (int n = 0; n < 4; ++n) accO[n][vv] *= alpha;
    }
    asm volatile("s_wait_dscnt 0x0" ::: "memory");  // per-wave Ps store->load

    // ---- O += P @ V (8 WMMA) ----
    Frag ap0, ap1;
    {
      int kg = hl * 16;
#pragma unroll
      for (int vv = 0; vv < 8; ++vv) {
        int k0 = kg + 2 * vv;  // K-dim = s_local
        ap0.u[vv] = *(const unsigned int*)&Ps[wid][ln][k0];
        ap1.u[vv] = *(const unsigned int*)&Ps[wid][ln][32 + k0];
      }
    }
#pragma unroll
    for (int nd = 0; nd < 4; ++nd) {
      Frag b0, b1;
      int dn = nd * 16 + ln;
      int k2 = hl * 16;
#pragma unroll
      for (int vv = 0; vv < 8; ++vv) {
        int ss = k2 + 2 * vv;                // K-dim = s, contiguous in Vt[d][s]
        b0.u[vv] = *(const unsigned int*)&Vt[dn][ss];
        b1.u[vv] = *(const unsigned int*)&Vt[dn][32 + ss];
      }
      accO[nd] = wmma_bf16(ap0.bf, b0.bf, accO[nd]);
      accO[nd] = wmma_bf16(ap1.bf, b1.bf, accO[nd]);
    }

    // ---- banded rel-v: O[t,d] += sum_j p[t, t-4+j] * relv[j][d] ----
    {
      int t0 = qBase + wid * 16;
      if (sBase <= t0 + 19 && sBase + 63 >= t0 - 4) {
#pragma unroll
        for (int nd = 0; nd < 4; ++nd)
#pragma unroll
          for (int vv = 0; vv < 8; ++vv) {
            int rw = hl * 8 + vv;
            int t  = t0 + rw;
            int dd = nd * 16 + ln;
#pragma unroll 1
            for (int j = 0; j < 9; ++j) {
              int sl = t - 4 + j - sBase;
              if (sl >= 0 && sl < 64)
                accO[nd][vv] += (float)Ps[wid][rw][sl] * relv[j][dd];
            }
          }
      }
    }
  }

  // ---- normalize and store bf16 [B,H,T,KC] ----
#pragma unroll
  for (int vv = 0; vv < 8; ++vv) {
    float inv = 1.f / lrun[vv];
    int t = qBase + wid * 16 + hl * 8 + vv;
#pragma unroll
    for (int nd = 0; nd < 4; ++nd) {
      int dd = nd * 16 + ln;
      att[((size_t)bh * T_ + t) * KC_ + dd] = (__bf16)(accO[nd][vv] * inv);
    }
  }
}

// ===========================================================================
// Kernel 3: output projection  out[b,o,t] = Wo[o,:]·att[b,:,t] + bo[o] (fp32)
// ===========================================================================
__global__ __launch_bounds__(128) void out_proj_kernel(
    const __bf16* __restrict__ att, const float* __restrict__ Wo,
    const float* __restrict__ bo, float* __restrict__ out) {
  __shared__ __bf16 Wl[64][36];
  __shared__ __bf16 XlT[64][40];   // [t][c]

  const int tTile = blockIdx.x, oTile = blockIdx.y, b = blockIdx.z;
  const int tid = threadIdx.x, lane = tid & 31, wid = tid >> 5;
  const int hl = lane >> 4, ln = lane & 15;
  const int wm = wid >> 1, wn = wid & 1;
  const int oBase = oTile * 64, tBase = tTile * 64;

  v8f acc[2][2];
#pragma unroll
  for (int i = 0; i < 2; ++i)
#pragma unroll
    for (int j = 0; j < 2; ++j) acc[i][j] = zero8();

#pragma unroll 1
  for (int kb = 0; kb < C_ / 32; ++kb) {
    for (int i = tid; i < 64 * 16; i += 128) {
      int r = i >> 4, p = i & 15;
      const float* wp = &Wo[(size_t)(oBase + r) * C_ + kb * 32 + 2 * p];
      ((unsigned int*)&Wl[r][0])[p] = pk_bf16(wp[0], wp[1]);
    }
    for (int i = tid; i < 32 * 64; i += 128) {
      int r = i >> 6, t = i & 63;
      int cidx = kb * 32 + r;
      int h = cidx >> 6, d = cidx & 63;
      XlT[t][r] = att[((size_t)(b * H_ + h) * T_ + tBase + t) * KC_ + d];
    }
    __syncthreads();
    if (kb + 1 < C_ / 32)
      __builtin_prefetch(&Wo[(size_t)(oBase + (tid >> 1)) * C_ + (kb + 1) * 32 + (tid & 1) * 16], 0, 0);

    Frag afr[2], bfr[2];
#pragma unroll
    for (int ms = 0; ms < 2; ++ms) {
      int mr = wm * 32 + ms * 16 + ln;
      int kg = hl * 16;
#pragma unroll
      for (int vv = 0; vv < 8; ++vv) {
        int k0 = kg + 2 * vv;
        afr[ms].u[vv] = *(const unsigned int*)&Wl[mr][k0];
      }
    }
#pragma unroll
    for (int ns = 0; ns < 2; ++ns) {
      int nc = wn * 32 + ns * 16 + ln;
      int k2 = hl * 16;
#pragma unroll
      for (int vv = 0; vv < 8; ++vv)
        bfr[ns].u[vv] = *(const unsigned int*)&XlT[nc][k2 + 2 * vv];
    }
#pragma unroll
    for (int ms = 0; ms < 2; ++ms)
#pragma unroll
      for (int ns = 0; ns < 2; ++ns)
        acc[ms][ns] = wmma_bf16(afr[ms].bf, bfr[ns].bf, acc[ms][ns]);
    __syncthreads();
  }

#pragma unroll
  for (int ms = 0; ms < 2; ++ms)
#pragma unroll
    for (int ns = 0; ns < 2; ++ns)
#pragma unroll
      for (int vv = 0; vv < 8; ++vv) {
        int o = oBase + wm * 32 + ms * 16 + hl * 8 + vv;
        int t = tBase + wn * 32 + ns * 16 + ln;
        out[((size_t)b * C_ + o) * T_ + t] = acc[ms][ns][vv] + bo[o];
      }
}

// ===========================================================================
extern "C" void kernel_launch(void* const* d_in, const int* in_sizes, int n_in,
                              void* d_out, int out_size, void* d_ws, size_t ws_size,
                              hipStream_t stream) {
  (void)in_sizes; (void)n_in; (void)out_size; (void)ws_size;
  const float* x    = (const float*)d_in[0];
  const float* c    = (const float*)d_in[1];
  const float* Wq   = (const float*)d_in[2];
  const float* bq   = (const float*)d_in[3];
  const float* Wk   = (const float*)d_in[4];
  const float* bk   = (const float*)d_in[5];
  const float* Wv   = (const float*)d_in[6];
  const float* bv   = (const float*)d_in[7];
  const float* Wo   = (const float*)d_in[8];
  const float* bo   = (const float*)d_in[9];
  const float* relk = (const float*)d_in[10];  // [1,9,64]
  const float* relv = (const float*)d_in[11];  // [1,9,64]
  float* out = (float*)d_out;

  const size_t nper = (size_t)B_ * H_ * T_ * KC_;  // 3,145,728 elems (bf16)
  __bf16* qws = (__bf16*)d_ws;
  __bf16* kws = qws + nper;
  __bf16* vws = kws + nper;
  __bf16* aws = vws + nper;  // 24 MB total workspace

  qkv_proj_kernel<<<dim3(T_ / 64, C_ / 64, 3 * B_), 128, 0, stream>>>(
      x, c, Wq, bq, Wk, bk, Wv, bv, qws, kws, vws);
  attn_kernel<<<dim3(T_ / 64, B_ * H_), 128, 0, stream>>>(
      qws, kws, vws, relk, relv, aws);
  out_proj_kernel<<<dim3(T_ / 64, C_ / 64, B_), 128, 0, stream>>>(
      aws, Wo, bo, out);
}
